// ADIAModel_74088185856594
// MI455X (gfx1250) — matrix operand order repaired and checked by reading, not compiled, not run on hardware
//
#include <hip/hip_runtime.h>
#include <hip/hip_bf16.h>
#include <math.h>

// ---- model constants ----
#define BB   32
#define NN   512
#define DD   64
#define CC   8
#define EE   132      // P*(P-1), P=12
#define P2N  10
#define NCIN 10

typedef __attribute__((ext_vector_type(16))) _Float16 v16h;
typedef __attribute__((ext_vector_type(8)))  _Float16 v8h;
typedef __attribute__((ext_vector_type(8)))  float    v8f;

__device__ __forceinline__ v16h cat8(v8h lo, v8h hi){
  return __builtin_shufflevector(lo,hi,0,1,2,3,4,5,6,7,8,9,10,11,12,13,14,15);
}
__device__ __forceinline__ v8f wmma16(v16h a, v16h b, v8f c){
  // D = A(16x32 f16) * B(32x16 f16) + C(16x16 f32)
  return __builtin_amdgcn_wmma_f32_16x16x32_f16(false,a,false,b,(short)0,c,false,false);
}
__device__ __forceinline__ float gelu_f(float x){
  return 0.5f*x*(1.0f + erff(x*0.70710678118654752f));
}

// =====================================================================
// Kernel 1: stem + 5 residual conv blocks (WMMA) + mean over N.
// One 8-wave block per edge; h kept resident in LDS as f16 [n][d] with
// zero halo columns for the k=3 SAME conv.
// =====================================================================
struct ExtractParams {
  const float* x;     // (B*E, C, N)
  const float* sw;    // (D, C) stem_w
  const float* sb;    // (D,)   stem_b
  const float* w[5];  // (D, D, 3)
  const float* g[5];  // (D,)
  const float* bb[5]; // (D,)
  float* emb;         // (B*E, D)
};

__global__ __launch_bounds__(256) void extract_kernel(ExtractParams P)
{
  extern __shared__ char sm_[];
  _Float16* hT = (_Float16*)sm_;          // (514, 64) f16, rows 0 & 513 = zero halo
  _Float16* yT = hT + 514*64;             // (512, 64) f16
  _Float16* wk = yT + 512*64;             // (3, 64, 64) f16, [tap][out][in]
  float* xs    = (float*)yT;              // alias: (8, 512) f32, stem stage only
  float* gsum  = (float*)(wk + 3*64*64);  // 8
  float* gsq   = gsum + 8;                // 8
  float* gmean = gsq + 8;                 // 8
  float* grstd = gmean + 8;               // 8
  float* part  = grstd + 8;               // 256

  const int blk  = blockIdx.x;
  const int tid  = threadIdx.x;
  const int wave = tid>>5, lane = tid&31, ln = lane&15, hs = lane>>4;

  // ---------------- stem: h[d][n] = sum_c sw[d][c]*x[c][n] + sb[d] --------
  for (int i=tid;i<CC*NN;i+=256) xs[i] = P.x[(size_t)blk*CC*NN + i];
  if (tid<64){ hT[tid]=(_Float16)0.f; hT[513*64+tid]=(_Float16)0.f; }
  __syncthreads();
  {
    int m = tid&63;
    float swr[CC];
    #pragma unroll
    for (int c=0;c<CC;c++) swr[c]=P.sw[m*CC+c];
    float sbm = P.sb[m];
    for (int j=0;j<128;j++){
      int n = (tid>>6) + 4*j;
      float a = sbm;
      #pragma unroll
      for (int c=0;c<CC;c++) a += swr[c]*xs[c*NN+n];
      hT[(n+1)*64+m] = (_Float16)a;
    }
  }
  __syncthreads();

  // ---------------- 5 residual conv blocks ----------------
  for (int l=0;l<5;l++){
    if (tid<8){ gsum[tid]=0.f; gsq[tid]=0.f; }
    const float* W = P.w[l];
    for (int i=tid;i<3*64*64;i+=256){
      int t=i>>12, rem=i&4095, m=rem>>6, k=rem&63;
      wk[i] = (_Float16)W[(m*64+k)*3 + t];      // (o,i,tap) -> [tap][o][i]
      if (l<4) __builtin_prefetch(P.w[l+1]+i, 0, 1);
    }
    __syncthreads();

    // y = conv3(h): wave owns 64 columns; tiles 16x16; K=64 per tap
    for (int mt=0;mt<4;mt++){
      v16h af[6];
      #pragma unroll
      for (int t=0;t<3;t++)
        #pragma unroll
        for (int kc=0;kc<2;kc++){
          const _Float16* ap = wk + t*4096 + (mt*16+ln)*64 + kc*32 + hs*8;
          af[t*2+kc] = cat8(*(const v8h*)ap, *(const v8h*)(ap+16));
        }
      for (int nt=0;nt<4;nt++){
        int ncol = wave*64 + nt*16 + ln;        // 0..511
        v8f acc;
        #pragma unroll
        for (int r=0;r<8;r++) acc[r]=0.f;
        #pragma unroll
        for (int t=0;t<3;t++){
          const _Float16* hrow = hT + (ncol+t)*64;   // halo handles n+t-1
          #pragma unroll
          for (int kc=0;kc<2;kc++){
            const _Float16* bp = hrow + kc*32 + hs*8;
            v16h bf = cat8(*(const v8h*)bp, *(const v8h*)(bp+16));
            acc = wmma16(af[t*2+kc], bf, acc);
          }
        }
        // store y (f16) + group-norm partial stats
        float s=0.f, sq=0.f;
        #pragma unroll
        for (int r=0;r<8;r++){
          float v=acc[r]; s+=v; sq+=v*v;
          yT[ncol*64 + mt*16 + hs*8 + r] = (_Float16)v;
        }
        #pragma unroll
        for (int o=8;o;o>>=1){ s += __shfl_xor(s,o,32); sq += __shfl_xor(sq,o,32); }
        if (ln==0){ atomicAdd(&gsum[2*mt+hs], s); atomicAdd(&gsq[2*mt+hs], sq); }
      }
    }
    __syncthreads();
    if (tid<8){
      float mu  = gsum[tid]*(1.f/4096.f);
      float var = gsq[tid]*(1.f/4096.f) - mu*mu;
      gmean[tid]=mu; grstd[tid]=rsqrtf(var+1e-5f);
    }
    __syncthreads();
    // h += gelu(groupnorm(y)*g + b)
    {
      int m=tid&63, grp=m>>3;
      float gam = P.g[l][m], bet = P.bb[l][m];
      float mu = gmean[grp], rs = grstd[grp];
      for (int j=0;j<128;j++){
        int n = (tid>>6) + 4*j;
        float y = (float)yT[n*64+m];
        float v = (y-mu)*rs*gam + bet;
        float hv = (float)hT[(n+1)*64+m] + gelu_f(v);
        hT[(n+1)*64+m] = (_Float16)hv;
      }
    }
    __syncthreads();
  }

  // ---------------- mean over N ----------------
  {
    int m=tid&63, q=tid>>6;
    float p=0.f;
    for (int j=0;j<128;j++) p += (float)hT[(q+4*j+1)*64+m];
    part[tid]=p;
  }
  __syncthreads();
  if (tid<64){
    float s = part[tid]+part[64+tid]+part[128+tid]+part[192+tid];
    P.emb[(size_t)blk*64+tid] = s*(1.f/512.f);
  }
}

// =====================================================================
// Kernel 2: generic WMMA GEMM  C[M,N] = A[M,K] @ W[N,K]^T + bias, opt GELU.
// 128-row stripe per block, 16 rows per wave; K padded to 32, N to 16.
// =====================================================================
template<int ACT>
__global__ __launch_bounds__(256) void gemm_bias_act(
  const float* __restrict__ A, const float* __restrict__ W,
  const float* __restrict__ bias, float* __restrict__ Co,
  int M, int K, int N, int Kp, int Np, int ldc)
{
  extern __shared__ char sm_[];
  _Float16* As = (_Float16*)sm_;          // 128*Kp
  _Float16* Ws = As + 128*Kp;             // Np*Kp
  const int tid = threadIdx.x;
  const int m0  = blockIdx.x*128;
  for (int i=tid;i<128*Kp;i+=256){
    int r=i/Kp, k=i-r*Kp; int gm=m0+r;
    As[i] = (gm<M && k<K) ? (_Float16)A[(size_t)gm*K+k] : (_Float16)0.f;
  }
  for (int i=tid;i<Np*Kp;i+=256){
    int n=i/Kp, k=i-n*Kp;
    Ws[i] = (n<N && k<K) ? (_Float16)W[(size_t)n*K+k] : (_Float16)0.f;
  }
  __syncthreads();
  const int wave=tid>>5, lane=tid&31, ln=lane&15, hs=lane>>4;
  const _Float16* arow = As + (wave*16 + ln)*Kp;
  const int nkc = Kp>>5, nnt = Np>>4;
  for (int nt=0; nt<nnt; ++nt){
    const _Float16* brow = Ws + (nt*16 + ln)*Kp;
    v8f acc;
    #pragma unroll
    for (int r=0;r<8;++r) acc[r]=0.f;
    for (int kc=0; kc<nkc; ++kc){
      const _Float16* ap = arow + kc*32 + hs*8;
      v16h af = cat8(*(const v8h*)ap, *(const v8h*)(ap+16));
      const _Float16* bp = brow + kc*32 + hs*8;
      v16h bf = cat8(*(const v8h*)bp, *(const v8h*)(bp+16));
      acc = wmma16(af, bf, acc);
    }
    int gn = nt*16 + ln;
    float bv = (bias && gn<N) ? bias[gn] : 0.f;
    #pragma unroll
    for (int r=0;r<8;++r){
      int gm = m0 + wave*16 + hs*8 + r;
      if (gm<M && gn<N){
        float v = acc[r] + bv;
        if (ACT==1) v = gelu_f(v);
        Co[(size_t)gm*ldc + gn] = v;
      }
    }
  }
}

// =====================================================================
// Kernel 3: LayerNorm over dim 64 (opt residual add, opt GELU).
// =====================================================================
__global__ __launch_bounds__(256) void ln_kernel(
  const float* __restrict__ in, const float* __restrict__ res,
  const float* __restrict__ g, const float* __restrict__ b,
  float* __restrict__ out, int R, int ostride, int act)
{
  __shared__ float red[256];
  int tid=threadIdx.x; int r = blockIdx.x*4 + (tid>>6); int c = tid&63;
  float v = 0.f;
  if (r < R){ v = in[(size_t)r*64+c]; if (res) v += res[(size_t)r*64+c]; }
  red[tid]=v; __syncthreads();
  for (int o=32;o;o>>=1){ if (c<o) red[tid]+=red[tid+o]; __syncthreads(); }
  float mean = red[(tid>>6)<<6]*(1.f/64.f);
  __syncthreads();
  float d = v-mean; red[tid]=d*d; __syncthreads();
  for (int o=32;o;o>>=1){ if (c<o) red[tid]+=red[tid+o]; __syncthreads(); }
  float var = red[(tid>>6)<<6]*(1.f/64.f);
  float o_ = d*rsqrtf(var+1e-5f)*g[c]+b[c];
  if (act) o_ = gelu_f(o_);
  if (r<R) out[(size_t)r*ostride + c] = o_;
}

// =====================================================================
// Kernel 4: edge attention (132 tokens, 4 heads x hd16) with struct bias.
// One block per (b, head); one wave per score row.
// =====================================================================
__global__ __launch_bounds__(256) void attn_scores_kernel(
  const float* __restrict__ Qb, const float* __restrict__ Kb,
  const float* __restrict__ Vb, const int* __restrict__ rel,
  const float* __restrict__ sb, const unsigned char* __restrict__ mask,
  float* __restrict__ AO)
{
  __shared__ float qh[EE*16], kh[EE*16], vh[EE*16];
  __shared__ float srow[8][EE];
  __shared__ unsigned char mk[EE];
  const int b = blockIdx.x >> 2, h = blockIdx.x & 3;
  const int tid = threadIdx.x;
  for (int i=tid;i<EE*16;i+=256){
    int e=i>>4, d=i&15;
    size_t src = ((size_t)(b*EE+e))*64 + h*16 + d;
    qh[i]=Qb[src]; kh[i]=Kb[src]; vh[i]=Vb[src];
  }
  for (int i=tid;i<EE;i+=256) mk[i]=mask[b*EE+i];
  __syncthreads();
  const int wave=tid>>5, lane=tid&31;
  const float NEG = -3.0e38f;
  for (int i=wave; i<EE; i+=8){
    const float* qr = qh + i*16;
    const int* rrow = rel + ((size_t)(b*EE+i))*EE;
    float lmax = NEG;
    for (int j=lane; j<EE; j+=32){
      float s=0.f; const float* kr = kh + j*16;
      #pragma unroll
      for (int d=0;d<16;d++) s += qr[d]*kr[d];
      s = s*0.25f + sb[rrow[j]*4 + h];
      if (!mk[j]) s = NEG;
      srow[wave][j]=s;
      lmax = fmaxf(lmax, s);
    }
    #pragma unroll
    for (int o=16;o;o>>=1) lmax = fmaxf(lmax, __shfl_xor(lmax,o,32));
    float lsum=0.f;
    for (int j=lane;j<EE;j+=32){
      float ex = mk[j] ? __expf(srow[wave][j]-lmax) : 0.f;
      srow[wave][j]=ex; lsum += ex;
    }
    #pragma unroll
    for (int o=16;o;o>>=1) lsum += __shfl_xor(lsum,o,32);
    float inv = (lsum>0.f) ? 1.f/lsum : 0.f;   // fully-masked row -> 0 (NaN->0)
    float acc[16];
    #pragma unroll
    for (int d=0;d<16;d++) acc[d]=0.f;
    for (int j=lane;j<EE;j+=32){
      float wv = srow[wave][j]*inv;
      const float* vr = vh + j*16;
      #pragma unroll
      for (int d=0;d<16;d++) acc[d] += wv*vr[d];
    }
    #pragma unroll
    for (int d=0;d<16;d++)
      #pragma unroll
      for (int o=16;o;o>>=1) acc[d] += __shfl_xor(acc[d],o,32);
    if (lane==0){
      float* orow = AO + ((size_t)(b*EE+i))*64 + h*16;
      #pragma unroll
      for (int d=0;d<16;d++) orow[d]=acc[d];
    }
  }
}

// =====================================================================
// Small VALU kernels
// =====================================================================
__global__ __launch_bounds__(256) void concat_merge_kernel(
  const float* emb, const float* type_emb, const int* etypes, float* MI)
{
  int i = blockIdx.x*256 + threadIdx.x;
  if (i >= BB*EE*128) return;
  int row = i>>7, d = i&127;
  MI[i] = (d<64) ? emb[(size_t)row*64+d] : type_emb[etypes[row]*64 + (d-64)];
}

__global__ __launch_bounds__(256) void edge_head_kernel(
  const float* X, const float* w, const float* bias, float* out)
{
  int i = blockIdx.x*256 + threadIdx.x;
  if (i >= BB*EE) return;
  const float* xr = X + (size_t)i*64;
  #pragma unroll
  for (int k=0;k<2;k++){
    float s = bias[k];
    for (int d=0;d<64;d++) s += xr[d]*w[k*64+d];
    out[i*2+k]=s;
  }
}

__global__ __launch_bounds__(256) void gather_e4_kernel(
  const float* X, const int* gidx, float* E4)
{
  int i = blockIdx.x*256 + threadIdx.x;
  if (i >= BB*P2N*4*64) return;
  int d = i&63, r = i>>6;
  int q = r&3, pr = r>>2;
  int p = pr % P2N, b = pr / P2N;
  int e = gidx[p*4+q];
  E4[i] = X[((size_t)(b*EE+e))*64 + d];
}

__global__ __launch_bounds__(64) void node_attn_kernel(
  const float* __restrict__ QKV, float* __restrict__ ATT)
{
  int bp = blockIdx.x;           // b*10+p
  int tid = threadIdx.x;
  if (tid < 16){
    int h = tid>>2, qi = tid&3;
    const float* base = QKV + (size_t)bp*4*192;
    const float* qv = base + qi*192 + h*16;
    float sc[4]; float mx=-3.0e38f;
    #pragma unroll
    for (int k=0;k<4;k++){
      const float* kv = base + k*192 + 64 + h*16;
      float s=0.f;
      #pragma unroll
      for (int d=0;d<16;d++) s += qv[d]*kv[d];
      s *= 0.25f; sc[k]=s; mx=fmaxf(mx,s);
    }
    float sum=0.f;
    #pragma unroll
    for (int k=0;k<4;k++){ sc[k]=__expf(sc[k]-mx); sum+=sc[k]; }
    float inv=1.f/sum;
    float* orow = ATT + ((size_t)bp*4+qi)*64 + h*16;
    #pragma unroll
    for (int d=0;d<16;d++){
      float a=0.f;
      #pragma unroll
      for (int k=0;k<4;k++) a += sc[k]*inv*base[k*192+128+h*16+d];
      orow[d]=a;
    }
  }
}

__global__ __launch_bounds__(256) void pool_kernel(const float* NLN, float* MC){
  int i = blockIdx.x*256+threadIdx.x;
  if (i >= BB*P2N*64) return;
  int row=i>>6, d=i&63;
  float s=0.f;
  #pragma unroll
  for (int q=0;q<4;q++) s += NLN[((size_t)row*4+q)*64+d];
  MC[(size_t)row*192 + d] = s*0.25f;
}

__global__ __launch_bounds__(256) void inter_kernel(const float* E4, float* OUT){
  int i = blockIdx.x*256+threadIdx.x;
  if (i >= BB*P2N*384) return;
  int row=i/384, j=i-row*384;
  int pi=j>>6, d=j&63;
  const int p1[6]={0,0,0,1,1,2}, p2[6]={1,2,3,2,3,3};
  const float* base = E4 + (size_t)row*4*64;
  OUT[i] = base[p1[pi]*64+d]*base[p2[pi]*64+d];
}

// =====================================================================
// Host driver
// =====================================================================
extern "C" void kernel_launch(void* const* d_in, const int* in_sizes, int n_in,
                              void* d_out, int out_size, void* d_ws, size_t ws_size,
                              hipStream_t stream)
{
  (void)in_sizes; (void)n_in; (void)out_size; (void)ws_size;
  auto F = [&](int i){ return (const float*)d_in[i]; };
  auto I = [&](int i){ return (const int*)d_in[i]; };

  const float* edge_data = F(0);               // (B,E,C,N) f32
  const int*   edge_types = I(1);              // (B,E) i32
  const unsigned char* edge_mask = (const unsigned char*)d_in[2]; // (B,E) bool
  const float* node_ci = F(3);                 // (B,P2,NCI) f32
  const int*   struct_rel_p = I(4);            // (B,E,E) i32
  const int*   gather_idx = I(5);              // (P2,4) i32

  // params flattened as a JAX pytree (sorted dict keys), base index 6
  const int PB = 6;
  // attn[lp]: bk0 bo1 bq2 bv3 ff1b4 ff1w5 ff2b6 ff2w7 n1b8 n1g9 n2b10 n2g11 sb12 wk13 wo14 wq15 wv16
  auto ATT = [&](int lp, int k){ return F(PB + lp*17 + k); };
  // blocks[l]: b0 g1 w2
  auto BLK = [&](int l, int k){ return F(PB+34 + l*3 + k); };
  const float* eh_b = F(PB+49); const float* eh_w = F(PB+50);
  const float* merge_b    = F(PB+51); const float* merge_beta = F(PB+52);
  const float* merge_g    = F(PB+53); const float* merge_w    = F(PB+54);
  // node: ci1b0 ci1w1 ci2b2 ci2w3 ci_b4 ci_g5 cl1b6 cl1w7 cl2b8 cl2w9 in_b10 in_w11
  //       ip1b12 ip1w13 ip2b14 ip2w15 ip_b16 ip_g17 ln_b18 ln_g19 mg_b20 mg_beta21
  //       mg_g22 mg_w23 out_b24 out_w25
  auto NDP = [&](int k){ return F(PB+55+k); };
  const float* stem_b = F(PB+81); const float* stem_w = F(PB+82);
  const float* type_emb = F(PB+83);

  // ---- workspace ----
  float* wp = (float*)d_ws;
  auto alloc=[&](size_t n){ float* r=wp; wp+=n; return r; };
  const int MEDGE = BB*EE;    // 4224   (renamed: M_E is a math.h macro!)
  float* emb = alloc((size_t)MEDGE*64);
  float* MI  = alloc((size_t)MEDGE*128);
  float* MM  = alloc((size_t)MEDGE*64);
  float* X   = alloc((size_t)MEDGE*64);
  float* Qb  = alloc((size_t)MEDGE*64);
  float* Kb  = alloc((size_t)MEDGE*64);
  float* Vb  = alloc((size_t)MEDGE*64);
  float* AO  = alloc((size_t)MEDGE*64);
  float* Ob  = alloc((size_t)MEDGE*64);
  float* X1  = alloc((size_t)MEDGE*64);
  float* F1  = alloc((size_t)MEDGE*256);
  float* F2  = alloc((size_t)MEDGE*64);
  const int M4 = BB*P2N*4;    // 1280
  float* E4   = alloc((size_t)M4*64);
  float* NQKV = alloc((size_t)M4*192);
  float* NATT = alloc((size_t)M4*64);
  float* NOP  = alloc((size_t)M4*64);
  float* NLN  = alloc((size_t)M4*64);
  const int MN = BB*P2N;      // 320
  float* MC   = alloc((size_t)MN*192);
  float* INT_ = alloc((size_t)MN*384);
  float* IP1  = alloc((size_t)MN*128);
  float* IP2  = alloc((size_t)MN*64);
  float* CI1  = alloc((size_t)MN*64);
  float* CI2  = alloc((size_t)MN*64);
  float* MG0  = alloc((size_t)MN*64);
  float* MGb  = alloc((size_t)MN*64);
  float* CL1  = alloc((size_t)MN*64);

  // allow >64KB dynamic LDS (deterministic, capture-safe host calls)
  const int SM_EXTRACT = 157056;             // 514*64*2 + 512*64*2 + 3*64*64*2 + 288*4
  (void)hipFuncSetAttribute(reinterpret_cast<const void*>(&extract_kernel),
        hipFuncAttributeMaxDynamicSharedMemorySize, SM_EXTRACT);
  (void)hipFuncSetAttribute(reinterpret_cast<const void*>(&gemm_bias_act<0>),
        hipFuncAttributeMaxDynamicSharedMemorySize, 196608);
  (void)hipFuncSetAttribute(reinterpret_cast<const void*>(&gemm_bias_act<1>),
        hipFuncAttributeMaxDynamicSharedMemorySize, 196608);

  auto gemm=[&](const float* A, const float* W, const float* bias, float* C,
                int M,int K,int N,int ldc,int act){
    int Kp=(K+31)&~31, Np=(N+15)&~15;
    size_t sm=((size_t)128*Kp + (size_t)Np*Kp)*2;
    dim3 gr((M+127)/128);
    if (act) gemm_bias_act<1><<<gr,256,sm,stream>>>(A,W,bias,C,M,K,N,Kp,Np,ldc);
    else     gemm_bias_act<0><<<gr,256,sm,stream>>>(A,W,bias,C,M,K,N,Kp,Np,ldc);
  };
  auto ln=[&](const float* in, const float* res, const float* g, const float* b,
              float* out, int R, int ostride, int act){
    ln_kernel<<<dim3((R+3)/4),256,0,stream>>>(in,res,g,b,out,R,ostride,act);
  };

  // ---- 1) extract: conv stack (dominant FLOPs, WMMA) ----
  ExtractParams EP;
  EP.x=edge_data; EP.sw=stem_w; EP.sb=stem_b;
  for (int l=0;l<5;l++){ EP.w[l]=BLK(l,2); EP.g[l]=BLK(l,1); EP.bb[l]=BLK(l,0); }
  EP.emb=emb;
  extract_kernel<<<dim3(MEDGE),256,SM_EXTRACT,stream>>>(EP);

  // ---- 2) merge: concat(emb,type_emb) -> linear -> LN -> gelu ----
  concat_merge_kernel<<<dim3((MEDGE*128+255)/256),256,0,stream>>>(emb,type_emb,edge_types,MI);
  gemm(MI, merge_w, merge_b, MM, MEDGE,128,64,64,0);
  ln(MM, nullptr, merge_g, merge_beta, X, MEDGE,64,1);

  // ---- 3) two edge-attention layers ----
  for (int lp=0;lp<2;lp++){
    gemm(X, ATT(lp,15), ATT(lp,2), Qb, MEDGE,64,64,64,0);   // wq,bq
    gemm(X, ATT(lp,13), ATT(lp,0), Kb, MEDGE,64,64,64,0);   // wk,bk
    gemm(X, ATT(lp,16), ATT(lp,3), Vb, MEDGE,64,64,64,0);   // wv,bv
    attn_scores_kernel<<<dim3(BB*4),256,0,stream>>>(Qb,Kb,Vb,struct_rel_p,
                                                    ATT(lp,12),edge_mask,AO);
    gemm(AO, ATT(lp,14), ATT(lp,1), Ob, MEDGE,64,64,64,0);  // wo,bo
    ln(Ob, X, ATT(lp,9), ATT(lp,8), X1, MEDGE,64,0);        // n1
    gemm(X1, ATT(lp,5), ATT(lp,4), F1, MEDGE,64,256,256,1); // ff1 + gelu
    gemm(F1, ATT(lp,7), ATT(lp,6), F2, MEDGE,256,64,64,0);  // ff2
    ln(F2, X1, ATT(lp,11), ATT(lp,10), X, MEDGE,64,0);      // n2
  }

  // ---- 4) edge head -> d_out[0 : B*E*2) ----
  edge_head_kernel<<<dim3((MEDGE+255)/256),256,0,stream>>>(X, eh_w, eh_b, (float*)d_out);

  // ---- 5) node head ----
  gather_e4_kernel<<<dim3((M4*64+255)/256),256,0,stream>>>(X, gather_idx, E4);
  gemm(E4, NDP(11), NDP(10), NQKV, M4,64,192,192,0);      // in_w,in_b
  node_attn_kernel<<<dim3(MN),64,0,stream>>>(NQKV, NATT);
  gemm(NATT, NDP(25), NDP(24), NOP, M4,64,64,64,0);       // out_w,out_b
  ln(NOP, E4, NDP(19), NDP(18), NLN, M4,64,0);            // ln_g,ln_b
  pool_kernel<<<dim3((MN*64+255)/256),256,0,stream>>>(NLN, MC);       // -> MC[:,0:64]
  inter_kernel<<<dim3((MN*384+255)/256),256,0,stream>>>(E4, INT_);
  gemm(INT_, NDP(13), NDP(12), IP1, MN,384,128,128,1);    // ip1 + gelu
  gemm(IP1,  NDP(15), NDP(14), IP2, MN,128,64,64,0);      // ip2
  ln(IP2, nullptr, NDP(17), NDP(16), MC+64, MN,192,0);    // -> MC[:,64:128]
  gemm(node_ci, NDP(1), NDP(0), CI1, MN,10,64,64,1);      // ci1 + gelu
  gemm(CI1, NDP(3), NDP(2), CI2, MN,64,64,64,0);          // ci2
  ln(CI2, nullptr, NDP(5), NDP(4), MC+128, MN,192,0);     // -> MC[:,128:192]
  gemm(MC, NDP(23), NDP(20), MG0, MN,192,64,64,0);        // mg_w,mg_b
  ln(MG0, nullptr, NDP(22), NDP(21), MGb, MN,64,1);       // mg LN + gelu
  gemm(MGb, NDP(7), NDP(6), CL1, MN,64,64,64,1);          // cl1 + gelu
  gemm(CL1, NDP(9), NDP(8), (float*)d_out + (size_t)BB*EE*2, MN,64,8,8,0); // cl2
}
